// RelGCN_64381559767254
// MI455X (gfx1250) — matrix operand order repaired and checked by reading, not compiled, not run on hardware
//
#include <hip/hip_runtime.h>
#include <hip/hip_bf16.h>

#define N_NODES 50000
#define E_EDGES 800000
#define DDIM    128
#define R_REL   8
#define EPAD    (E_EDGES + 64 * R_REL)   // buckets padded to multiples of 64 edges
#define ASTRIDE 136                      // padded LDS row stride (bf16 elems) -> bank-conflict free

typedef __attribute__((ext_vector_type(16))) __bf16 v16bf;
typedef __attribute__((ext_vector_type(8)))  __bf16 v8bf;
typedef __attribute__((ext_vector_type(8)))  float  v8f;

// ---------------------------------------------------------------- helpers

__device__ __forceinline__ void atomicAddF(float* p, float v) {
  __hip_atomic_fetch_add(p, v, __ATOMIC_RELAXED, __HIP_MEMORY_SCOPE_AGENT);
}

// load a 16-element bf16 fragment from two (possibly disjoint) 16B-aligned runs
__device__ __forceinline__ v16bf ldfrag2(const __bf16* p1, const __bf16* p2) {
  v8bf a = *(const v8bf*)p1;
  v8bf b = *(const v8bf*)p2;
  return __builtin_shufflevector(a, b, 0,1,2,3,4,5,6,7,8,9,10,11,12,13,14,15);
}

// Stage one 128x128 fp32 weight matrix into LDS, transposed (Wt[n][k]) and
// split hi/lo bf16 so B fragments are contiguous 16-elem runs per lane.
__device__ __forceinline__ void stage_W(const float* __restrict__ Wg,
                                        __bf16* Whi, __bf16* Wlo, int tid) {
  const float4* Wg4 = (const float4*)Wg;
  #pragma unroll
  for (int it = 0; it < 32; ++it) {
    int idx = tid + it * 128;        // 4096 float4s total
    int k   = idx >> 5;              // source row (K)
    int n4  = (idx & 31) << 2;       // source col base (N)
    float4 w = Wg4[idx];
    float f[4] = {w.x, w.y, w.z, w.w};
    #pragma unroll
    for (int c = 0; c < 4; ++c) {
      int n = n4 + c;
      __bf16 h = (__bf16)f[c];
      __bf16 l = (__bf16)(f[c] - (float)h);
      Whi[n * ASTRIDE + k] = h;
      Wlo[n * ASTRIDE + k] = l;
    }
  }
}

// Stage 16 dense rows [rowbase, rowbase+16) of `in` into A hi/lo LDS (one wave).
template <bool RELU>
__device__ __forceinline__ void stage_A_dense(const float* __restrict__ in, int rowbase,
                                              __bf16* Ahi, __bf16* Alo, int lane) {
  int row   = lane >> 1;
  int cbase = (lane & 1) * 16;       // float4 index base within row
  const float4* rp = (const float4*)(in + (size_t)(rowbase + row) * DDIM);
  #pragma unroll
  for (int i = 0; i < 16; ++i) {
    int c4 = cbase + i;
    float4 w = rp[c4];
    float f[4] = {w.x, w.y, w.z, w.w};
    #pragma unroll
    for (int c = 0; c < 4; ++c) {
      float fv = RELU ? fmaxf(f[c], 0.0f) : f[c];
      __bf16 h = (__bf16)fv;
      Ahi[row * ASTRIDE + c4 * 4 + c] = h;
      Alo[row * ASTRIDE + c4 * 4 + c] = (__bf16)(fv - (float)h);
    }
  }
}

// Stage 16 gathered rows in[src[eids[m]]] (eids may contain -1 padding).
template <bool RELU>
__device__ __forceinline__ void stage_A_gather(const float* __restrict__ in,
                                               const int* __restrict__ eids,
                                               const int* __restrict__ srcA,
                                               __bf16* Ahi, __bf16* Alo, int lane) {
  int row   = lane >> 1;
  int cbase = (lane & 1) * 16;
  int ei = eids[row];
  const float4* rp = (ei >= 0) ? (const float4*)(in + (size_t)srcA[ei] * DDIM) : nullptr;
  #pragma unroll
  for (int i = 0; i < 16; ++i) {
    int c4 = cbase + i;
    float4 w = rp ? rp[c4] : make_float4(0.f, 0.f, 0.f, 0.f);
    float f[4] = {w.x, w.y, w.z, w.w};
    #pragma unroll
    for (int c = 0; c < 4; ++c) {
      float fv = RELU ? fmaxf(f[c], 0.0f) : f[c];
      __bf16 h = (__bf16)fv;
      Ahi[row * ASTRIDE + c4 * 4 + c] = h;
      Alo[row * ASTRIDE + c4 * 4 + c] = (__bf16)(fv - (float)h);
    }
  }
}

// 16x128 = (16x128) @ (128x128) tile via bf16x3 WMMA (hi*hi + hi*lo + lo*hi).
__device__ __forceinline__ void tile_gemm(const __bf16* Ahi, const __bf16* Alo,
                                          const __bf16* Whi, const __bf16* Wlo,
                                          int lane, v8f* acc) {
  const int half = lane >> 4;
  const int mrow = lane & 15;
  const __bf16* arh = Ahi + mrow * ASTRIDE;
  const __bf16* arl = Alo + mrow * ASTRIDE;
  #pragma unroll
  for (int k0 = 0; k0 < 128; k0 += 32) {
    // A fragment: K run [k0+16h, +16) contiguous (matches B)
    v16bf ah = ldfrag2(arh + k0 + 16 * half, arh + k0 + 16 * half + 8);
    v16bf al = ldfrag2(arl + k0 + 16 * half, arl + k0 + 16 * half + 8);
    #pragma unroll
    for (int nt = 0; nt < 8; ++nt) {
      // B fragment: lane n holds K = k0+16h .. +16 contiguously in transposed LDS
      const __bf16* bhp = Whi + (size_t)(nt * 16 + mrow) * ASTRIDE + k0 + 16 * half;
      const __bf16* blp = Wlo + (size_t)(nt * 16 + mrow) * ASTRIDE + k0 + 16 * half;
      v16bf bh = ldfrag2(bhp, bhp + 8);
      v16bf bl = ldfrag2(blp, blp + 8);
      acc[nt] = __builtin_amdgcn_wmma_f32_16x16x32_bf16(false, ah, false, bh, (short)0, acc[nt], false, false);
      acc[nt] = __builtin_amdgcn_wmma_f32_16x16x32_bf16(false, ah, false, bl, (short)0, acc[nt], false, false);
      acc[nt] = __builtin_amdgcn_wmma_f32_16x16x32_bf16(false, al, false, bh, (short)0, acc[nt], false, false);
    }
  }
}

// ---------------------------------------------------------------- bucketing

__global__ void k_zero(int* counts) {
  if (threadIdx.x < R_REL) counts[threadIdx.x] = 0;
}

__global__ __launch_bounds__(256) void k_hist(const int* __restrict__ et, int* counts) {
  __shared__ int lc[R_REL];
  if (threadIdx.x < R_REL) lc[threadIdx.x] = 0;
  __syncthreads();
  int e = blockIdx.x * 256 + threadIdx.x;
  if (e < E_EDGES) atomicAdd(&lc[et[e]], 1);
  __syncthreads();
  if (threadIdx.x < R_REL) atomicAdd(&counts[threadIdx.x], lc[threadIdx.x]);
}

__global__ void k_scan(const int* __restrict__ counts, int* offsets, int* cursor) {
  if (threadIdx.x == 0 && blockIdx.x == 0) {
    int off = 0;
    for (int r = 0; r < R_REL; ++r) {
      offsets[r] = off;
      cursor[r]  = off;
      off += ((counts[r] + 63) >> 6) << 6;   // 64-edge aligned buckets
    }
    offsets[R_REL] = off;
  }
}

__global__ __launch_bounds__(256) void k_fill(int* edge_ids) {
  int i = blockIdx.x * 256 + threadIdx.x;
  if (i < EPAD) edge_ids[i] = -1;
}

__global__ __launch_bounds__(256) void k_bucket(const int* __restrict__ et,
                                                int* cursor, int* edge_ids) {
  int e = blockIdx.x * 256 + threadIdx.x;
  if (e < E_EDGES) {
    int pos = atomicAdd(&cursor[et[e]], 1);
    edge_ids[pos] = e;
  }
}

// ---------------------------------------------------------------- GEMM kernels

// out[n,:] = (RELU? relu(in) : in)[n,:] @ loopW + b   (dense self-loop + bias)
template <bool RELU_IN>
__global__ __launch_bounds__(128) void k_loop(const float* __restrict__ in,
                                              const float* __restrict__ loopW,
                                              const float* __restrict__ bias,
                                              float* __restrict__ out) {
  __shared__ __align__(16) __bf16 Whi[128 * ASTRIDE];
  __shared__ __align__(16) __bf16 Wlo[128 * ASTRIDE];
  __shared__ __align__(16) __bf16 Ahi[4][16 * ASTRIDE];
  __shared__ __align__(16) __bf16 Alo[4][16 * ASTRIDE];

  int tid = threadIdx.x, wave = tid >> 5, lane = tid & 31;
  stage_W(loopW, Whi, Wlo, tid);

  int tile = blockIdx.x * 4 + wave;          // 16-node tile; N divisible by 16
  int rowbase = tile * 16;
  bool active = rowbase < N_NODES;
  if (active) stage_A_dense<RELU_IN>(in, rowbase, Ahi[wave], Alo[wave], lane);
  __syncthreads();
  if (!active) return;                       // wave-uniform

  int half = lane >> 4, col = lane & 15;
  v8f acc[8];
  #pragma unroll
  for (int nt = 0; nt < 8; ++nt) {
    float b = bias[nt * 16 + col];
    acc[nt] = (v8f){b, b, b, b, b, b, b, b};
  }

  tile_gemm(Ahi[wave], Alo[wave], Whi, Wlo, lane, acc);

  #pragma unroll
  for (int nt = 0; nt < 8; ++nt)
    #pragma unroll
    for (int j = 0; j < 8; ++j) {
      int m = j + 8 * half;                  // C/D layout: VGPR j -> M=j (+8 for hi lanes)
      out[(size_t)(rowbase + m) * DDIM + nt * 16 + col] = acc[nt][j];
    }
}

// For each bucketed edge tile: msg = in[src] @ W[etype]; atomically add to out[dst].
template <bool RELU_IN>
__global__ __launch_bounds__(128) void k_edge(const float* __restrict__ in,
                                              const float* __restrict__ Wg,
                                              const int* __restrict__ srcA,
                                              const int* __restrict__ dstA,
                                              const int* __restrict__ et,
                                              const int* __restrict__ edge_ids,
                                              const int* __restrict__ offsets,
                                              float* __restrict__ out) {
  __shared__ __align__(16) __bf16 Whi[128 * ASTRIDE];
  __shared__ __align__(16) __bf16 Wlo[128 * ASTRIDE];
  __shared__ __align__(16) __bf16 Ahi[4][16 * ASTRIDE];
  __shared__ __align__(16) __bf16 Alo[4][16 * ASTRIDE];
  __shared__ int dstrow[4][16];

  int tid = threadIdx.x, wave = tid >> 5, lane = tid & 31;
  int base = blockIdx.x * 64;                // 64 edges per workgroup (4 tiles)
  int total = offsets[R_REL];
  if (base >= total) return;                 // block-uniform

  int e0 = edge_ids[base];                   // first edge of bucket-aligned block: valid
  int r  = et[e0];
  stage_W(Wg + (size_t)r * DDIM * DDIM, Whi, Wlo, tid);

  const int* eids = edge_ids + base + wave * 16;
  stage_A_gather<RELU_IN>(in, eids, srcA, Ahi[wave], Alo[wave], lane);
  if (lane < 16) {
    int ei = eids[lane];
    dstrow[wave][lane] = (ei >= 0) ? dstA[ei] : -1;
  }
  __syncthreads();

  v8f acc[8] = {};
  tile_gemm(Ahi[wave], Alo[wave], Whi, Wlo, lane, acc);

  int half = lane >> 4, col = lane & 15;
  #pragma unroll
  for (int nt = 0; nt < 8; ++nt)
    #pragma unroll
    for (int j = 0; j < 8; ++j) {
      int m = j + 8 * half;
      int dr = dstrow[wave][m];
      if (dr >= 0) atomicAddF(&out[(size_t)dr * DDIM + nt * 16 + col], acc[nt][j]);
    }
}

// ---------------------------------------------------------------- L2 normalize

__global__ __launch_bounds__(256) void k_norm(float* __restrict__ out) {
  int wave = threadIdx.x >> 5, lane = threadIdx.x & 31;
  int row = blockIdx.x * 8 + wave;
  if (row >= N_NODES) return;
  float4* p = (float4*)(out + (size_t)row * DDIM) + lane;
  float4 v = *p;
  float s = v.x * v.x + v.y * v.y + v.z * v.z + v.w * v.w;
  #pragma unroll
  for (int off = 16; off > 0; off >>= 1) s += __shfl_xor(s, off, 32);
  float inv = 1.0f / fmaxf(sqrtf(s), 1e-12f);
  v.x *= inv; v.y *= inv; v.z *= inv; v.w *= inv;
  *p = v;
}

// ---------------------------------------------------------------- launch

extern "C" void kernel_launch(void* const* d_in, const int* in_sizes, int n_in,
                              void* d_out, int out_size, void* d_ws, size_t ws_size,
                              hipStream_t stream) {
  (void)in_sizes; (void)n_in; (void)out_size; (void)ws_size;

  const float* x   = (const float*)d_in[0];
  const float* W1  = (const float*)d_in[1];
  const float* lW1 = (const float*)d_in[2];
  const float* b1  = (const float*)d_in[3];
  const float* W2  = (const float*)d_in[4];
  const float* lW2 = (const float*)d_in[5];
  const float* b2  = (const float*)d_in[6];
  const int* srcA  = (const int*)d_in[7];
  const int* dstA  = (const int*)d_in[8];
  const int* et    = (const int*)d_in[9];
  float* out = (float*)d_out;

  // workspace carve-up (~28.9 MB)
  char* p = (char*)d_ws;
  float* h1 = (float*)p;                 p += (size_t)N_NODES * DDIM * sizeof(float);
  int* counts   = (int*)p;               p += R_REL * sizeof(int);
  int* offsets  = (int*)p;               p += (R_REL + 1) * sizeof(int);
  int* cursor   = (int*)p;               p += R_REL * sizeof(int);
  int* edge_ids = (int*)p;               // EPAD ints

  const int histBlocks = (E_EDGES + 255) / 256;
  const int fillBlocks = (EPAD + 255) / 256;
  const int loopBlocks = (N_NODES / 16 + 3) / 4;   // 3125 tiles -> 782 blocks
  const int edgeBlocks = EPAD / 64;                // 12508

  // Bucket edges by relation (counting sort into 64-aligned buckets)
  k_zero<<<1, 32, 0, stream>>>(counts);
  k_hist<<<histBlocks, 256, 0, stream>>>(et, counts);
  k_scan<<<1, 1, 0, stream>>>(counts, offsets, cursor);
  k_fill<<<fillBlocks, 256, 0, stream>>>(edge_ids);
  k_bucket<<<histBlocks, 256, 0, stream>>>(et, cursor, edge_ids);

  // Layer 1: h1 = x@loopW1 + b1 ; h1 += scatter-add(x[src] @ W1[etype])
  k_loop<false><<<loopBlocks, 128, 0, stream>>>(x, lW1, b1, h1);
  k_edge<false><<<edgeBlocks, 128, 0, stream>>>(x, W1, srcA, dstA, et, edge_ids, offsets, h1);

  // Layer 2 (ReLU fused into operand staging): out = relu(h1)@loopW2 + b2 + agg
  k_loop<true><<<loopBlocks, 128, 0, stream>>>(h1, lW2, b2, out);
  k_edge<true><<<edgeBlocks, 128, 0, stream>>>(h1, W2, srcA, dstA, et, edge_ids, offsets, out);

  // Row-wise L2 normalize
  k_norm<<<N_NODES / 8, 256, 0, stream>>>(out);
}
